// GraphConvolutionLayer_84954453115174
// MI455X (gfx1250) — matrix operand order repaired and checked by reading, not compiled, not run on hardware
//
#include <hip/hip_runtime.h>
#include <hip/hip_bf16.h>
#include <math.h>

// ---------------------------------------------------------------------------
// Problem constants (match reference): B=8, N=4096, H=256, K=32
// ---------------------------------------------------------------------------
#define BATCH   8
#define NNODES  4096
#define HD      256
#define TOPK    32
#define NEGINF  (-3.0e38f)
#define LDA     264   // padded LDS row stride (elements): 132 dwords -> 4-bank shift/row

typedef __attribute__((ext_vector_type(16))) __bf16 bf16x16;
typedef __attribute__((ext_vector_type(8)))  __bf16 bf16x8;
typedef __attribute__((ext_vector_type(4)))  __bf16 bf16x4;
typedef __attribute__((ext_vector_type(8)))  float  f32x8;
typedef __attribute__((ext_vector_type(4)))  float  f32x4;

// ---------------------------------------------------------------------------
// WMMA fragment loaders (wave32, v_wmma_f32_16x16x32_bf16)
//
// A (16x32, MxK, row-major, row stride = ld elements):
//   lanes 0-15 : row M = lane,    K octets {kb..kb+7}, {kb+16..kb+23}, kb=0
//   lanes 16-31: row M = lane-16, kb=8
// B (32x16, KxN) loaded from B^T stored row-major [n][k]:
//   lanes 0-15 : col N = lane,    K octets {kb..kb+7}, {kb+16..kb+23}, kb=0
//   lanes 16-31: col N = lane-16, kb=8
// C/D (16x16 f32): vgpr r, lane l -> M = r + 8*(l>>4), N = l&15
// ---------------------------------------------------------------------------
__device__ __forceinline__ bf16x16 load_frag_a(const __bf16* tile, int ld, int lane) {
  const int row = lane & 15;
  const int kb  = (lane >> 4) * 8;
  union { bf16x16 v; bf16x8 h[2]; } u;
  u.h[0] = *(const bf16x8*)(tile + row * ld + kb);
  u.h[1] = *(const bf16x8*)(tile + row * ld + kb + 16);
  return u.v;
}

__device__ __forceinline__ bf16x16 load_frag_b(const __bf16* tileT, int ld, int lane) {
  const int col = lane & 15;
  const int kb  = (lane >> 4) * 8;
  union { bf16x16 v; bf16x8 h[2]; } u;
  u.h[0] = *(const bf16x8*)(tileT + col * ld + kb);
  u.h[1] = *(const bf16x8*)(tileT + col * ld + kb + 16);
  return u.v;
}

#define WMMA_BF16(a, b, c) \
  __builtin_amdgcn_wmma_f32_16x16x32_bf16(false, (a), false, (b), (short)0, (c), false, false)

// ---------------------------------------------------------------------------
// gfx1250 async global->LDS copy (32 bytes per call) + wait.
// A flat pointer's low 32 bits are the wave-relative LDS byte offset
// (ISA 10.2); the async op's VDST operand is the
// LDS byte address. Tracked by ASYNCcnt.
// ---------------------------------------------------------------------------
__device__ __forceinline__ void async_copy32(const __bf16* lds_dst, const __bf16* gsrc) {
  uint32_t loff = (uint32_t)(uintptr_t)lds_dst;
  uint64_t g    = (uint64_t)(uintptr_t)gsrc;
  asm volatile("global_load_async_to_lds_b128 %0, %1, off\n\t"
               "global_load_async_to_lds_b128 %2, %3, off"
               :: "v"(loff), "v"(g), "v"(loff + 16u), "v"(g + 16ull)
               : "memory");
}
__device__ __forceinline__ void wait_async0() {
  asm volatile("s_wait_asynccnt 0" ::: "memory");
}

// WGP-scope prefetch (scope field 0): pulls the line into ALL cache levels
// including the WGP cache (ISA 10.5), unlike SE/DEV scope which stop at GL2.
// Covers 256B starting at p.
__device__ __forceinline__ void prefetch_wgp256(const void* p) {
  uint64_t a = (uint64_t)(uintptr_t)p;
  asm volatile("global_prefetch_b8 %0, off\n\t"
               "global_prefetch_b8 %0, off offset:128"
               :: "v"(a));
}

// ---------------------------------------------------------------------------
// Kernel 0: f32 -> bf16 conversion (nodevecs, weights)
// ---------------------------------------------------------------------------
__global__ __launch_bounds__(256) void cvt_bf16_kernel(const float* __restrict__ in,
                                                       __bf16* __restrict__ out, int n) {
  int i = blockIdx.x * 256 + threadIdx.x;
  if (i < n) out[i] = (__bf16)in[i];
}

// ---------------------------------------------------------------------------
// Kernel 1: adj = nv1 @ nv2^T (bf16 WMMA), streaming top-32 + softmax per row.
// One workgroup (8 waves) per 16 adjacency rows; adj is never materialized.
// Wave owns rows 2w, 2w+1; top-32 list lives in registers (one entry/lane),
// merged per 128-column chunk by a wave-argmax tournament vs LDS scores.
// ---------------------------------------------------------------------------
__device__ __forceinline__ void merge_row(volatile float* srow, int colbase,
                                          float& bv, int& bi, int lane) {
  for (;;) {
    float mv = NEGINF; int mp = lane;
#pragma unroll
    for (int j = 0; j < 4; ++j) {
      float c = srow[lane + 32 * j];
      if (c > mv) { mv = c; mp = lane + 32 * j; }
    }
#pragma unroll
    for (int off = 16; off > 0; off >>= 1) {
      float ov = __shfl_xor(mv, off, 32);
      int   op = __shfl_xor(mp, off, 32);
      if (ov > mv || (ov == mv && op < mp)) { mv = ov; mp = op; }
    }
    float lmv = bv; int lml = lane;
#pragma unroll
    for (int off = 16; off > 0; off >>= 1) {
      float ov = __shfl_xor(lmv, off, 32);
      int   ol = __shfl_xor(lml, off, 32);
      if (ov < lmv || (ov == lmv && ol < lml)) { lmv = ov; lml = ol; }
    }
    if (mv <= lmv) break;                 // chunk exhausted vs list minimum
    if (lane == lml) { bv = mv; bi = colbase + mp; }
    if (lane == 0) srow[mp] = NEGINF;     // same-wave LDS ops stay in order
  }
}

__global__ __launch_bounds__(256) void adj_topk_kernel(const __bf16* __restrict__ nv1b,
                                                       const __bf16* __restrict__ nv2b,
                                                       float* __restrict__ probs,
                                                       int* __restrict__ idxs) {
  __shared__ float scores[16][128];
  const int lane = threadIdx.x & 31;
  const int wave = threadIdx.x >> 5;
  const int row0 = blockIdx.x * 16;

  // A fragments for the whole K=256 reduction, kept in registers
  bf16x16 afr[8];
#pragma unroll
  for (int ks = 0; ks < 8; ++ks)
    afr[ks] = load_frag_a(nv1b + row0 * HD + ks * 32, HD, lane);

  // Warm the WGP cache for chunk 0's B tile before the first compute phase.
  prefetch_wgp256(nv2b + wave * 16 * HD + lane * 128);

  float bv0 = NEGINF, bv1 = NEGINF;
  int   bi0 = 0,      bi1 = 0;
  const int r0 = wave * 2, r1 = wave * 2 + 1;

  for (int chunk = 0; chunk < NNODES / 128; ++chunk) {
    const int colbase = chunk * 128;
    const __bf16* btile = nv2b + (colbase + wave * 16) * HD;
    // Two independent accumulation chains (even/odd k-steps).
    f32x8 acca = {}, accb = {};
#pragma unroll
    for (int ks = 0; ks < 8; ks += 2) {
      bf16x16 ba = load_frag_b(btile + ks * 32, HD, lane);
      bf16x16 bb = load_frag_b(btile + (ks + 1) * 32, HD, lane);
      acca = WMMA_BF16(afr[ks], ba, acca);
      accb = WMMA_BF16(afr[ks + 1], bb, accb);
    }
    const f32x8 acc = acca + accb;
#pragma unroll
    for (int r = 0; r < 8; ++r)
      scores[r + 8 * (lane >> 4)][wave * 16 + (lane & 15)] = acc[r];
    __syncthreads();
    // WGP-scope prefetch of next chunk's B tile (8KB/wave) so the line is in
    // the near cache by the time the serialized load->wmma chain runs again;
    // latency overlaps the merge phase below.
    if (chunk + 1 < NNODES / 128) {
      const __bf16* nb = nv2b + (colbase + 128 + wave * 16) * HD;
      prefetch_wgp256(nb + lane * 128);
    }
    merge_row(&scores[r0][0], colbase, bv0, bi0, lane);
    merge_row(&scores[r1][0], colbase, bv1, bi1, lane);
    __syncthreads();
  }

  // softmax over the 32 surviving values of each row (one value per lane)
#pragma unroll
  for (int rr = 0; rr < 2; ++rr) {
    float v = rr ? bv1 : bv0;
    int   i = rr ? bi1 : bi0;
    float mx = v;
#pragma unroll
    for (int off = 16; off > 0; off >>= 1) mx = fmaxf(mx, __shfl_xor(mx, off, 32));
    float e = __expf(v - mx);
    float s = e;
#pragma unroll
    for (int off = 16; off > 0; off >>= 1) s += __shfl_xor(s, off, 32);
    const int row = row0 + wave * 2 + rr;
    probs[row * TOPK + lane] = e / s;
    idxs [row * TOPK + lane] = i;
  }
}

// ---------------------------------------------------------------------------
// Kernel 2: gc gather (32 weighted neighbor rows) + residual + LayerNorm1.
// One wave per (b, m) row; lane owns two float4 slots (h = 4*lane, 128+4*lane)
// so every access is a b128 and each instruction covers 512B contiguous.
// ---------------------------------------------------------------------------
__global__ __launch_bounds__(256) void gather_ln1_kernel(const float* __restrict__ src,
                                                         const float* __restrict__ tgt,
                                                         const float* __restrict__ probs,
                                                         const int* __restrict__ idxs,
                                                         const float* __restrict__ ln1w,
                                                         const float* __restrict__ ln1b,
                                                         float* __restrict__ tgt1f,
                                                         __bf16* __restrict__ tgt1b) {
  const int lane = threadIdx.x & 31;
  const int wave = threadIdx.x >> 5;
  const int row  = blockIdx.x * 8 + wave;      // 0 .. B*N-1
  const int b    = row >> 12;
  const int m    = row & (NNODES - 1);
  const int h0   = lane * 4;
  const int h1   = 128 + lane * 4;

  const float pk = probs[m * TOPK + lane];
  const int   nk = idxs [m * TOPK + lane];

  f32x4 acc0 = {0.f, 0.f, 0.f, 0.f};
  f32x4 acc1 = {0.f, 0.f, 0.f, 0.f};
  for (int k = 0; k < TOPK; ++k) {
    const float p = __shfl(pk, k, 32);
    const int   n = __shfl(nk, k, 32);
    const float* s = src + (b * NNODES + n) * HD;
    acc0 += (*(const f32x4*)(s + h0)) * p;
    acc1 += (*(const f32x4*)(s + h1)) * p;
  }

  const float* t = tgt + row * HD;
  const f32x4 x0 = (*(const f32x4*)(t + h0)) + acc0;
  const f32x4 x1 = (*(const f32x4*)(t + h1)) + acc1;
  float s1 = 0.f, s2 = 0.f;
#pragma unroll
  for (int c = 0; c < 4; ++c) {
    s1 += x0[c] + x1[c];
    s2 += x0[c] * x0[c] + x1[c] * x1[c];
  }
#pragma unroll
  for (int off = 16; off > 0; off >>= 1) {
    s1 += __shfl_xor(s1, off, 32);
    s2 += __shfl_xor(s2, off, 32);
  }
  const float mu   = s1 * (1.f / HD);
  const float var  = s2 * (1.f / HD) - mu * mu;
  const float rstd = rsqrtf(var + 1e-5f);

  const f32x4 w0 = *(const f32x4*)(ln1w + h0);
  const f32x4 w1 = *(const f32x4*)(ln1w + h1);
  const f32x4 g0 = *(const f32x4*)(ln1b + h0);
  const f32x4 g1 = *(const f32x4*)(ln1b + h1);
  const f32x4 y0 = (x0 - mu) * rstd * w0 + g0;
  const f32x4 y1 = (x1 - mu) * rstd * w1 + g1;
  *(f32x4*)(tgt1f + row * HD + h0) = y0;
  *(f32x4*)(tgt1f + row * HD + h1) = y1;
  bf16x4 z0, z1;
#pragma unroll
  for (int c = 0; c < 4; ++c) { z0[c] = (__bf16)y0[c]; z1[c] = (__bf16)y1[c]; }
  *(bf16x4*)(tgt1b + row * HD + h0) = z0;
  *(bf16x4*)(tgt1b + row * HD + h1) = z1;
}

// ---------------------------------------------------------------------------
// Shared helper: stage one 16x256 bf16 A tile into padded LDS via async DMA.
// 256 threads x 32B = 8KB. Row stride LDA=264 elements kills the 16-way
// bank conflict of the fragment reads (132 dwords -> 4-bank shift per row).
// ---------------------------------------------------------------------------
__device__ __forceinline__ void stage_a_tile(__bf16* aLds, const __bf16* aglob, int tid) {
  const int row = tid >> 4;        // 0..15
  const int seg = tid & 15;        // 16 x 32B segments per 512B row
  async_copy32(aLds + row * LDA + seg * 16, aglob + row * HD + seg * 16);
  wait_async0();
  __syncthreads();
}

// ---------------------------------------------------------------------------
// Kernel 3: FF1 = relu(tgt1 @ w1^T + b1) -> bf16. Block owns 16 rows x 256
// cols; A tile staged once in LDS (async), wave w computes cols [32w,32w+32).
// ---------------------------------------------------------------------------
__global__ __launch_bounds__(256) void ff1_kernel(const __bf16* __restrict__ xb,
                                                  const __bf16* __restrict__ w1b,
                                                  const float* __restrict__ b1,
                                                  __bf16* __restrict__ x1b) {
  __shared__ __bf16 aLds[16 * LDA];
  const int lane = threadIdx.x & 31;
  const int wave = threadIdx.x >> 5;
  const int m0 = blockIdx.x * 16;
  const int n0 = wave * 32;

  stage_a_tile(aLds, xb + m0 * HD, threadIdx.x);

  f32x8 acc0 = {}, acc1 = {};
#pragma unroll
  for (int ks = 0; ks < 8; ++ks) {
    bf16x16 bA = load_frag_b(w1b + n0 * HD + ks * 32, HD, lane);
    bf16x16 bB = load_frag_b(w1b + (n0 + 16) * HD + ks * 32, HD, lane);
    bf16x16 a  = load_frag_a(aLds + ks * 32, LDA, lane);   // ds_load from LDS
    acc0 = WMMA_BF16(a, bA, acc0);
    acc1 = WMMA_BF16(a, bB, acc1);
  }

  const int nA = n0 + (lane & 15), nB = nA + 16;
  const float biasA = b1[nA], biasB = b1[nB];
#pragma unroll
  for (int r = 0; r < 8; ++r) {
    const int m = m0 + r + 8 * (lane >> 4);
    const float vA = acc0[r] + biasA;
    const float vB = acc1[r] + biasB;
    x1b[m * HD + nA] = (__bf16)(vA > 0.f ? vA : 0.f);
    x1b[m * HD + nB] = (__bf16)(vB > 0.f ? vB : 0.f);
  }
}

// ---------------------------------------------------------------------------
// Kernel 4: FF2 + bias + residual + LayerNorm2 -> out (f32). Same tiling as
// FF1; z staged in LDS, then row-wise LN with shuffle reductions (float4).
// ---------------------------------------------------------------------------
__global__ __launch_bounds__(256) void ff2_ln2_kernel(const __bf16* __restrict__ x1b,
                                                      const __bf16* __restrict__ w2b,
                                                      const float* __restrict__ b2,
                                                      const float* __restrict__ tgt1f,
                                                      const float* __restrict__ ln2w,
                                                      const float* __restrict__ ln2b,
                                                      float* __restrict__ out) {
  __shared__ __bf16 aLds[16 * LDA];
  __shared__ float zbuf[16][HD];
  const int lane = threadIdx.x & 31;
  const int wave = threadIdx.x >> 5;
  const int m0 = blockIdx.x * 16;
  const int n0 = wave * 32;

  stage_a_tile(aLds, x1b + m0 * HD, threadIdx.x);

  f32x8 acc0 = {}, acc1 = {};
#pragma unroll
  for (int ks = 0; ks < 8; ++ks) {
    bf16x16 bA = load_frag_b(w2b + n0 * HD + ks * 32, HD, lane);
    bf16x16 bB = load_frag_b(w2b + (n0 + 16) * HD + ks * 32, HD, lane);
    bf16x16 a  = load_frag_a(aLds + ks * 32, LDA, lane);
    acc0 = WMMA_BF16(a, bA, acc0);
    acc1 = WMMA_BF16(a, bB, acc1);
  }

  const int nA = n0 + (lane & 15), nB = nA + 16;
  const float biasA = b2[nA], biasB = b2[nB];
#pragma unroll
  for (int r = 0; r < 8; ++r) {
    const int ml = r + 8 * (lane >> 4);
    zbuf[ml][nA] = acc0[r] + biasA;
    zbuf[ml][nB] = acc1[r] + biasB;
  }
  __syncthreads();

  const int h0 = lane * 4;
  const int h1 = 128 + lane * 4;
#pragma unroll
  for (int rr = 0; rr < 2; ++rr) {
    const int ml  = wave * 2 + rr;
    const int row = m0 + ml;
    const f32x4 x0 = (*(const f32x4*)&zbuf[ml][h0]) + (*(const f32x4*)(tgt1f + row * HD + h0));
    const f32x4 x1 = (*(const f32x4*)&zbuf[ml][h1]) + (*(const f32x4*)(tgt1f + row * HD + h1));
    float s1 = 0.f, s2 = 0.f;
#pragma unroll
    for (int c = 0; c < 4; ++c) {
      s1 += x0[c] + x1[c];
      s2 += x0[c] * x0[c] + x1[c] * x1[c];
    }
#pragma unroll
    for (int off = 16; off > 0; off >>= 1) {
      s1 += __shfl_xor(s1, off, 32);
      s2 += __shfl_xor(s2, off, 32);
    }
    const float mu   = s1 * (1.f / HD);
    const float var  = s2 * (1.f / HD) - mu * mu;
    const float rstd = rsqrtf(var + 1e-5f);
    const f32x4 w0 = *(const f32x4*)(ln2w + h0);
    const f32x4 w1 = *(const f32x4*)(ln2w + h1);
    const f32x4 g0 = *(const f32x4*)(ln2b + h0);
    const f32x4 g1 = *(const f32x4*)(ln2b + h1);
    *(f32x4*)(out + row * HD + h0) = (x0 - mu) * rstd * w0 + g0;
    *(f32x4*)(out + row * HD + h1) = (x1 - mu) * rstd * w1 + g1;
  }
}

// ---------------------------------------------------------------------------
// Host launcher
// ---------------------------------------------------------------------------
extern "C" void kernel_launch(void* const* d_in, const int* in_sizes, int n_in,
                              void* d_out, int out_size, void* d_ws, size_t ws_size,
                              hipStream_t stream) {
  (void)in_sizes; (void)n_in; (void)out_size; (void)ws_size;
  const float* src  = (const float*)d_in[0];
  const float* tgt  = (const float*)d_in[1];
  const float* nv1  = (const float*)d_in[2];
  const float* nv2  = (const float*)d_in[3];
  const float* w1   = (const float*)d_in[4];
  const float* b1   = (const float*)d_in[5];
  const float* w2   = (const float*)d_in[6];
  const float* b2   = (const float*)d_in[7];
  const float* ln1w = (const float*)d_in[8];
  const float* ln1b = (const float*)d_in[9];
  const float* ln2w = (const float*)d_in[10];
  const float* ln2b = (const float*)d_in[11];
  float* out = (float*)d_out;

  char* ws = (char*)d_ws;
  size_t off = 0;
  auto carve = [&](size_t bytes) -> char* {
    char* p = ws + off;
    off = (off + bytes + 255) & ~(size_t)255;
    return p;
  };
  __bf16* nv1b  = (__bf16*)carve((size_t)NNODES * HD * sizeof(__bf16));
  __bf16* nv2b  = (__bf16*)carve((size_t)NNODES * HD * sizeof(__bf16));
  __bf16* w1b   = (__bf16*)carve((size_t)HD * HD * sizeof(__bf16));
  __bf16* w2b   = (__bf16*)carve((size_t)HD * HD * sizeof(__bf16));
  float*  probs = (float*) carve((size_t)NNODES * TOPK * sizeof(float));
  int*    idxs  = (int*)   carve((size_t)NNODES * TOPK * sizeof(int));
  float*  tgt1f = (float*) carve((size_t)BATCH * NNODES * HD * sizeof(float));
  __bf16* tgt1b = (__bf16*)carve((size_t)BATCH * NNODES * HD * sizeof(__bf16));
  __bf16* x1b   = (__bf16*)carve((size_t)BATCH * NNODES * HD * sizeof(__bf16));

  const int nvElems = NNODES * HD;  // 1,048,576
  const int wElems  = HD * HD;      // 65,536
  cvt_bf16_kernel<<<(nvElems + 255) / 256, 256, 0, stream>>>(nv1, nv1b, nvElems);
  cvt_bf16_kernel<<<(nvElems + 255) / 256, 256, 0, stream>>>(nv2, nv2b, nvElems);
  cvt_bf16_kernel<<<(wElems  + 255) / 256, 256, 0, stream>>>(w1,  w1b,  wElems);
  cvt_bf16_kernel<<<(wElems  + 255) / 256, 256, 0, stream>>>(w2,  w2b,  wElems);

  adj_topk_kernel<<<NNODES / 16, 256, 0, stream>>>(nv1b, nv2b, probs, idxs);

  gather_ln1_kernel<<<(BATCH * NNODES) / 8, 256, 0, stream>>>(
      src, tgt, probs, idxs, ln1w, ln1b, tgt1f, tgt1b);

  ff1_kernel<<<(BATCH * NNODES) / 16, 256, 0, stream>>>(tgt1b, w1b, b1, x1b);

  ff2_ln2_kernel<<<(BATCH * NNODES) / 16, 256, 0, stream>>>(
      x1b, w2b, b2, tgt1f, ln2w, ln2b, out);
}
